// model_24507083391623
// MI455X (gfx1250) — matrix operand order repaired and checked by reading, not compiled, not run on hardware
//
#include <hip/hip_runtime.h>
#include <hip/hip_bf16.h>
#include <math.h>

typedef __attribute__((ext_vector_type(16))) _Float16 v16h;
typedef __attribute__((ext_vector_type(8)))  _Float16 v8h;
typedef __attribute__((ext_vector_type(8)))  float    v8f;

#define B_   2
#define C_   128
#define HW_  16384          // H*W
#define N_   32768          // B*H*W
#define E_   6
#define D_   6
#define HF_  512
#define EPS_LOG 2.220446049250313e-16f

#if __has_builtin(__builtin_amdgcn_global_load_async_to_lds_b128) && \
    __has_builtin(__builtin_amdgcn_s_wait_asynccnt)
#define HAVE_ASYNC_LDS 1
#else
#define HAVE_ASYNC_LDS 0
#endif

#if HAVE_ASYNC_LDS
// Builtin signature (leaked by clang diagnostic): param0 is
// 'int __vector_size__(16) __device__ *' (AS1 global), param1 the AS3 (LDS)
// counterpart, then imm offset + imm cpol.
typedef int v4i_vs __attribute__((vector_size(16)));
typedef __attribute__((address_space(1))) v4i_vs as1_v4i;
typedef __attribute__((address_space(3))) v4i_vs as3_v4i;
// global VA is the AS1 pointer value; low 32 bits of a generic LDS pointer are
// the LDS byte offset.
__device__ __forceinline__ void async_cp_b128(const void* g, void* l) {
    __builtin_amdgcn_global_load_async_to_lds_b128(
        (as1_v4i*)(uintptr_t)g,
        (as3_v4i*)(uint32_t)(uintptr_t)l, 0, 0);
}
#define WAIT_ASYNC(n) __builtin_amdgcn_s_wait_asynccnt(n)
#else
#define WAIT_ASYNC(n)
#endif

__device__ __forceinline__ v16h cat8(v8h lo, v8h hi) {
    v16h r;
#pragma unroll
    for (int i = 0; i < 8; ++i) { r[i] = lo[i]; r[i + 8] = hi[i]; }
    return r;
}

// ---------------------------------------------------------------------------
// prep: per-batch gating constants + zero global accumulators (every call)
// addl[b][e] = prompt[b]@w_g[C:2C, e] + gate_boost*(de_cls[b]@degra_w + degra_b)[e]
// ---------------------------------------------------------------------------
__global__ void prep_kernel(const float* __restrict__ prompt,
                            const float* __restrict__ de_cls,
                            const float* __restrict__ w_g,
                            const float* __restrict__ gate_boost,
                            const float* __restrict__ degra_w,
                            const float* __restrict__ degra_b,
                            float* __restrict__ addl,
                            float* __restrict__ Weight,
                            float* __restrict__ Load) {
    const int t = threadIdx.x;
    if (t < B_ * E_) {
        const int b = t / E_, e = t % E_;
        float p = 0.f;
        for (int c = 0; c < C_; ++c)
            p += prompt[b * C_ + c] * w_g[(C_ + c) * E_ + e];
        float bias = degra_b[e];
        for (int d = 0; d < D_; ++d)
            bias += de_cls[b * D_ + d] * degra_w[d * E_ + e];
        addl[t] = p + gate_boost[0] * bias;
    }
    if (t < E_) { Weight[t] = 0.f; Load[t] = 0.f; }
}

// ---------------------------------------------------------------------------
// convw: fp32 -> f16 transpose of expert weights so LDS fills are contiguous.
// W1t[(e*HF+hf)*C + c] = fc1_w[e][c][hf]   (per-output-column contiguous in K)
// W2t[(e*C +c )*HF+ hf] = fc2_w[e][hf][c]
// ---------------------------------------------------------------------------
__global__ __launch_bounds__(256) void convw_kernel(const float* __restrict__ fc1_w,
                                                    const float* __restrict__ fc2_w,
                                                    _Float16* __restrict__ W1t,
                                                    _Float16* __restrict__ W2t) {
    const int idx = blockIdx.x * 256 + threadIdx.x;   // E*HF*C total
    {
        const int c  = idx % C_;
        const int hf = (idx / C_) % HF_;
        const int e  = idx / (C_ * HF_);
        W1t[idx] = (_Float16)fc1_w[((e * C_) + c) * HF_ + hf];
    }
    {
        const int hf = idx % HF_;
        const int c  = (idx / HF_) % C_;
        const int e  = idx / (HF_ * C_);
        W2t[idx] = (_Float16)fc2_w[((e * HF_) + hf) * C_ + c];
    }
}

// ---------------------------------------------------------------------------
// gate: per-token logits, top-2 softmax gates, fused x -> f16 conversion.
// ---------------------------------------------------------------------------
__global__ __launch_bounds__(256) void gate_kernel(const float* __restrict__ x,
                                                   const float* __restrict__ w_g,
                                                   const float* __restrict__ addl,
                                                   _Float16* __restrict__ Xh,
                                                   float* __restrict__ wgates,
                                                   float* __restrict__ Weight,
                                                   float* __restrict__ Load) {
    __shared__ float sWg[C_ * E_];
    const int tid = threadIdx.x;
    for (int i = tid; i < C_ * E_; i += 256) sWg[i] = w_g[i];
    __syncthreads();

    const int n  = blockIdx.x * 256 + tid;
    const int b  = n >> 14;            // n / (H*W)
    const int hw = n & (HW_ - 1);

    float logit[E_];
#pragma unroll
    for (int e = 0; e < E_; ++e) logit[e] = addl[b * E_ + e];

    const float* xb  = x + (size_t)b * C_ * HW_ + hw;
    _Float16*    xhr = Xh + (size_t)n * C_;
    for (int c = 0; c < C_; ++c) {
        const float v = xb[(size_t)c * HW_];
        xhr[c] = (_Float16)v;
#pragma unroll
        for (int e = 0; e < E_; ++e) logit[e] += v * sWg[c * E_ + e];
    }

    // top-2 (ties -> lowest index, matching lax.top_k)
    float v1 = -INFINITY; int i1 = 0;
#pragma unroll
    for (int e = 0; e < E_; ++e) if (logit[e] > v1) { v1 = logit[e]; i1 = e; }
    float v2 = -INFINITY; int i2 = 0;
#pragma unroll
    for (int e = 0; e < E_; ++e) if (e != i1 && logit[e] > v2) { v2 = logit[e]; i2 = e; }

    const float ex = __expf(v2 - v1);          // v2 <= v1 -> stable
    const float g1 = 1.f / (1.f + ex);
    const float g2 = ex / (1.f + ex);

    float* wr = wgates + (size_t)n * E_;
#pragma unroll
    for (int e = 0; e < E_; ++e) wr[e] = 0.f;
    wr[i1] = g1; wr[i2] = g2;

    atomicAdd(&Weight[i1], g1); atomicAdd(&Weight[i2], g2);
    atomicAdd(&Load[i1], 1.f);  atomicAdd(&Load[i2], 1.f);
}

// ---------------------------------------------------------------------------
// loss: balance(Weight) + balance(load), var with ddof=1 over E=6.
// ---------------------------------------------------------------------------
__global__ void loss_kernel(const float* __restrict__ Weight,
                            const float* __restrict__ Load,
                            float* __restrict__ out_loss) {
    if (threadIdx.x != 0) return;
    float total = 0.f;
    const float* arr[2] = { Weight, Load };
#pragma unroll
    for (int a = 0; a < 2; ++a) {
        float m = 0.f;
        for (int e = 0; e < E_; ++e) m += arr[a][e];
        m *= (1.f / E_);
        float v = 0.f;
        for (int e = 0; e < E_; ++e) { const float d = arr[a][e] - m; v += d * d; }
        v *= (1.f / (E_ - 1));
        total += v / (m * m + 1e-10f);
    }
    *out_loss = total;
}

// ---------------------------------------------------------------------------
// moe: fused all-expert MLP with WMMA f32_16x16x32_f16.
// Block = 8 waves; wave w owns tokens [tile*128 + 16w, +16).
// Weight chunks double-buffered in LDS via GLOBAL_LOAD_ASYNC_TO_LDS_B128
// (ASYNCcnt-tracked; next chunk in flight while current chunk computes).
// comb accumulated in registers across all 6 experts -> no output atomics.
// ---------------------------------------------------------------------------
__global__ __launch_bounds__(256) void moe_kernel(const _Float16* __restrict__ Xh,
                                                  const _Float16* __restrict__ W1t,
                                                  const _Float16* __restrict__ W2t,
                                                  const float* __restrict__ fc1_b,
                                                  const float* __restrict__ fc2_b,
                                                  const float* __restrict__ wg,
                                                  float* __restrict__ out) {
    __shared__ _Float16 sW1[2][32 * C_];    // [buf][col(hf)][k=C]   2 x 8 KB
    __shared__ _Float16 sW2[2][C_ * 32];    // [buf][col(c)][k(hf)]  2 x 8 KB
    __shared__ _Float16 sHid[8][16 * 32];   // per-wave D->A relayout      8 KB

    const int tid  = threadIdx.x;
    const int wave = tid >> 5;
    const int lane = tid & 31;
    const int l16  = lane & 15;
    const int lhi  = lane >> 4;             // 0: low half, 1: high half
    const int tok0 = blockIdx.x * 128 + wave * 16;

    // X A-tiles (16x128, ISA A-layout): lane = row (mod 16); per 16x32 K-tile,
    // lane<16 holds K = [k0..k0+7] ++ [k0+16..k0+23], lane>=16 is +8 shifted.
    v16h ax[4];
    {
        const _Float16* xr = Xh + (size_t)(tok0 + l16) * C_;
#pragma unroll
        for (int kk = 0; kk < 4; ++kk) {
            const int s0 = kk * 32 + (lhi ? 8 : 0);
            const v8h lo = *reinterpret_cast<const v8h*>(xr + s0);
            const v8h hi = *reinterpret_cast<const v8h*>(xr + s0 + 16);
            ax[kk] = cat8(lo, hi);
        }
    }

    // per-thread staging of one (expert, hf-chunk) weight slice: 4 x b128
    auto stage = [&](int it, int buf) {
        const int e   = it >> 4;
        const int hf0 = (it & 15) * 32;
        const int col1 = tid >> 3;            // 0..31
        const int kp1  = (tid & 7) * 16;      // 0..112
        const _Float16* s1 = W1t + (size_t)(e * HF_ + hf0 + col1) * C_ + kp1;
        _Float16*       d1 = &sW1[buf][col1 * C_ + kp1];
        const int col2 = tid >> 1;            // 0..127
        const int kp2  = (tid & 1) * 16;      // 0 or 16
        const _Float16* s2 = W2t + (size_t)(e * C_ + col2) * HF_ + hf0 + kp2;
        _Float16*       d2 = &sW2[buf][col2 * 32 + kp2];
#if HAVE_ASYNC_LDS
        async_cp_b128(s1, d1);     async_cp_b128(s1 + 8, d1 + 8);
        async_cp_b128(s2, d2);     async_cp_b128(s2 + 8, d2 + 8);
#else
        *reinterpret_cast<v8h*>(d1)     = *reinterpret_cast<const v8h*>(s1);
        *reinterpret_cast<v8h*>(d1 + 8) = *reinterpret_cast<const v8h*>(s1 + 8);
        *reinterpret_cast<v8h*>(d2)     = *reinterpret_cast<const v8h*>(s2);
        *reinterpret_cast<v8h*>(d2 + 8) = *reinterpret_cast<const v8h*>(s2 + 8);
#endif
    };

    const v8f vzero = {};
    v8f comb[8];
#pragma unroll
    for (int t = 0; t < 8; ++t) comb[t] = vzero;
    v8f acc[8];
#pragma unroll
    for (int t = 0; t < 8; ++t) acc[t] = vzero;

    constexpr int TOT = E_ * (HF_ / 32);    // 96 chunk-iterations
    stage(0, 0);

    for (int it = 0; it < TOT; ++it) {
        const int e   = it >> 4;
        const int hf0 = (it & 15) * 32;
        const int buf = it & 1;

        __syncthreads();                     // all waves done reading buf^1
        if (it + 1 < TOT) { stage(it + 1, buf ^ 1); WAIT_ASYNC(4); }
        else              { WAIT_ASYNC(0); }
        __syncthreads();                     // chunk `it` resident in LDS

        if ((it & 15) == 0) {
#pragma unroll
            for (int t = 0; t < 8; ++t) acc[t] = vzero;
        }

        // fc1: Hid(16x32) = X(16x128) @ W1chunk(128x32), 8 WMMAs
        v8f hid[2] = { vzero, vzero };
#pragma unroll
        for (int kk = 0; kk < 4; ++kk) {
#pragma unroll
            for (int tc = 0; tc < 2; ++tc) {
                const _Float16* bp = &sW1[buf][(tc * 16 + l16) * C_ + kk * 32 + (lhi ? 8 : 0)];
                const v16h bm = cat8(*reinterpret_cast<const v8h*>(bp),
                                     *reinterpret_cast<const v8h*>(bp + 16));
                hid[tc] = __builtin_amdgcn_wmma_f32_16x16x32_f16(
                    false, ax[kk], false, bm, (short)0, hid[tc], false, false);
            }
        }

        // bias + exact gelu, D-layout -> per-wave LDS (row-major 16x32)
        _Float16* hbase = sHid[wave];
#pragma unroll
        for (int tc = 0; tc < 2; ++tc) {
            const float b1 = fc1_b[e * HF_ + hf0 + tc * 16 + l16];
#pragma unroll
            for (int i = 0; i < 8; ++i) {
                float v = hid[tc][i] + b1;
                v = 0.5f * v * (1.0f + erff(v * 0.70710678118654752f));
                hbase[(i + 8 * lhi) * 32 + tc * 16 + l16] = (_Float16)v;
            }
        }
        // re-read in A-layout (wave-private region; DS ops in-order per wave)
        const _Float16* hr = hbase + l16 * 32 + (lhi ? 8 : 0);
        const v16h ah = cat8(*reinterpret_cast<const v8h*>(hr),
                             *reinterpret_cast<const v8h*>(hr + 16));

        // fc2: Out(16x128) += Hid(16x32) @ W2chunk(32x128), 8 WMMAs
#pragma unroll
        for (int t = 0; t < 8; ++t) {
            const _Float16* bp = &sW2[buf][(t * 16 + l16) * 32 + (lhi ? 8 : 0)];
            const v16h bm = cat8(*reinterpret_cast<const v8h*>(bp),
                                 *reinterpret_cast<const v8h*>(bp + 16));
            acc[t] = __builtin_amdgcn_wmma_f32_16x16x32_f16(
                false, ah, false, bm, (short)0, acc[t], false, false);
        }

        if ((it & 15) == 15) {
            // combine for expert e: comb[n,c] += gate[n,e] * exp(out + b2)
            float g[8];
#pragma unroll
            for (int i = 0; i < 8; ++i)
                g[i] = wg[(size_t)(tok0 + i + 8 * lhi) * E_ + e];
#pragma unroll
            for (int t = 0; t < 8; ++t) {
                const float b2 = fc2_b[e * C_ + t * 16 + l16];
#pragma unroll
                for (int i = 0; i < 8; ++i)
                    comb[t][i] += g[i] * __expf(acc[t][i] + b2);
            }
        }
    }

    // y = log(comb, zeros->eps), scatter to NCHW
#pragma unroll
    for (int t = 0; t < 8; ++t) {
        const int c = t * 16 + l16;
#pragma unroll
        for (int i = 0; i < 8; ++i) {
            const int token = tok0 + i + 8 * lhi;
            const int b  = token >> 14;
            const int hw = token & (HW_ - 1);
            float v = comb[t][i];
            v = (v == 0.f) ? EPS_LOG : v;
            out[((size_t)(b * C_ + c)) * HW_ + hw] = __logf(v);
        }
    }
}

// ---------------------------------------------------------------------------
extern "C" void kernel_launch(void* const* d_in, const int* in_sizes, int n_in,
                              void* d_out, int out_size, void* d_ws, size_t ws_size,
                              hipStream_t stream) {
    const float* x          = (const float*)d_in[0];
    const float* prompt     = (const float*)d_in[1];
    const float* de_cls     = (const float*)d_in[2];
    const float* w_g        = (const float*)d_in[3];
    const float* gate_boost = (const float*)d_in[4];
    const float* degra_w    = (const float*)d_in[5];
    const float* degra_b    = (const float*)d_in[6];
    const float* fc1_w      = (const float*)d_in[7];
    const float* fc1_b      = (const float*)d_in[8];
    const float* fc2_w      = (const float*)d_in[9];
    const float* fc2_b      = (const float*)d_in[10];

    float* y    = (float*)d_out;              // [B,C,H,W] flat
    float* loss = y + (size_t)B_ * C_ * HW_;  // +1 scalar

    char* ws = (char*)d_ws;
    float*    wgates = (float*)(ws + 0);                       // N*E f32   786432 B
    float*    addl   = (float*)(ws + 786432);                  // B*E
    float*    Weight = (float*)(ws + 786944);                  // E
    float*    Load   = (float*)(ws + 787200);                  // E
    _Float16* Xh     = (_Float16*)(ws + 787456);               // N*C f16   8 MB
    _Float16* W1t    = (_Float16*)(ws + 787456 + 8388608);     // E*HF*C f16
    _Float16* W2t    = (_Float16*)(ws + 787456 + 8388608 + 786432);

    prep_kernel<<<1, 32, 0, stream>>>(prompt, de_cls, w_g, gate_boost,
                                      degra_w, degra_b, addl, Weight, Load);
    convw_kernel<<<(E_ * C_ * HF_) / 256, 256, 0, stream>>>(fc1_w, fc2_w, W1t, W2t);
    gate_kernel<<<N_ / 256, 256, 0, stream>>>(x, w_g, addl, Xh, wgates, Weight, Load);
    loss_kernel<<<1, 32, 0, stream>>>(Weight, Load, loss);
    moe_kernel<<<N_ / 128, 256, 0, stream>>>(Xh, W1t, W2t, fc1_b, fc2_b, wgates, y);
}
